// PointNetEncoder_62105227100296
// MI455X (gfx1250) — matrix operand order repaired and checked by reading, not compiled, not run on hardware
//
#include <hip/hip_runtime.h>
#include <math.h>

// ---------------- types ----------------
typedef _Float16 half_t;
typedef __attribute__((ext_vector_type(16))) _Float16 v16h;
typedef __attribute__((ext_vector_type(8)))  _Float16 v8h;
typedef __attribute__((ext_vector_type(8)))  float    v8f;

#define NEG_INF (-__builtin_inff())

// Problem constants (match reference)
#define BATCH 8
#define NPTS  2048
#define KNN   64
#define S1    1024
#define S2    256

// ---------------- WMMA fragment loaders ----------------
// A operand (16x32 f16), row-major source, row length `ld` halves (multiple of 8, 16B-aligned rows).
// ISA 16-bit A layout: lane L (m=L&15, hi=L>>4): halves 0..7 <-> k=kb+8*hi+{0..7},
// halves 8..15 <-> k=kb+16+8*hi+{0..7}  (two 16B contiguous runs).
__device__ __forceinline__ v16h load_frag_a(const half_t* base, int ld, int row0, int kb) {
    int lane = threadIdx.x & 31;
    int r = lane & 15, hi = lane >> 4;
    const half_t* p = base + (size_t)(row0 + r) * (size_t)ld + kb + 8 * hi;
    v8h lo = *(const v8h*)(p);
    v8h hh = *(const v8h*)(p + 16);
    v16h out;
#pragma unroll
    for (int i = 0; i < 8; i++) { out[i] = lo[i]; out[i + 8] = hh[i]; }
    return out;
}

// B operand (32x16 f16) built from Wt stored row-major [Cout][Kp] (i.e. B^T).
// ISA 16-bit B layout (same striping as A): lane L (n=L&15, hi=L>>4)
// halves 0..7 <-> k=kb+8*hi+{0..7}, halves 8..15 <-> k=kb+16+8*hi+{0..7} (two 16B runs).
__device__ __forceinline__ v16h load_frag_b(const half_t* base, int ld, int n0, int kb) {
    int lane = threadIdx.x & 31;
    int n = lane & 15, hi = lane >> 4;
    const half_t* p = base + (size_t)(n0 + n) * (size_t)ld + kb + 8 * hi;
    v8h lo = *(const v8h*)(p);
    v8h hh = *(const v8h*)(p + 16);
    v16h out;
#pragma unroll
    for (int i = 0; i < 8; i++) { out[i] = lo[i]; out[i + 8] = hh[i]; }
    return out;
}

__device__ __forceinline__ v8f wmma_f16(v16h a, v16h b, v8f c) {
    return __builtin_amdgcn_wmma_f32_16x16x32_f16(false, a, false, b, (short)0, c, false, false);
}

__device__ __forceinline__ float atomicMaxF(float* addr, float val) {
    int* ia = (int*)addr;
    int old = *ia;
    while (__int_as_float(old) < val) {
        int assumed = old;
        old = atomicCAS(ia, assumed, __float_as_int(val));
        if (old == assumed) break;
    }
    return __int_as_float(old);
}

// Warm L2/WGP caches with a weight buffer (lowers to global_prefetch_b8).
__device__ __forceinline__ void prefetch_buf(const half_t* p, int halves) {
    for (int o = (int)threadIdx.x * 32; o < halves; o += 256 * 32)
        __builtin_prefetch((const void*)(p + o), 0, 1);
}

// ---------------- small utility kernels ----------------
__global__ void wconv_kernel(const float* __restrict__ W, int Cin, int Cout, int Kp,
                             half_t* __restrict__ Wt) {
    int total = Cout * Kp;
    for (int i = blockIdx.x * blockDim.x + threadIdx.x; i < total; i += gridDim.x * blockDim.x) {
        int o = i / Kp, k = i % Kp;
        float v = (k < Cin) ? W[(size_t)k * Cout + o] : 0.0f;
        Wt[i] = (half_t)v;
    }
}

__global__ void init_out_kernel(float* out, int n) {
    for (int i = blockIdx.x * blockDim.x + threadIdx.x; i < n; i += gridDim.x * blockDim.x)
        out[i] = NEG_INF;
}

__global__ void build_a3_kernel(const float* __restrict__ h2, const float* __restrict__ cpos2,
                                half_t* __restrict__ A3) {
    const int KP = 288, C = 256, total = BATCH * S2 * KP;
    for (int i = blockIdx.x * blockDim.x + threadIdx.x; i < total; i += gridDim.x * blockDim.x) {
        int row = i / KP, c = i % KP;
        float v = 0.0f;
        if (c < C) v = h2[(size_t)row * C + c];
        else if (c < C + 3) v = cpos2[(size_t)row * 3 + (c - C)];
        A3[i] = (half_t)v;
    }
}

// ---------------- FPS: one workgroup per cloud ----------------
__global__ __launch_bounds__(256) void fps_kernel(const float* __restrict__ pos, int Np, int S,
                                                  float* __restrict__ cpos_out) {
    extern __shared__ float sm[];
    float* px = sm;
    float* py = px + Np;
    float* pz = py + Np;
    float* dd = pz + Np;
    float* rv = dd + Np;            // 256
    int*   ri = (int*)(rv + 256);   // 256
    int cloud = blockIdx.x;
    int tid = threadIdx.x;
    const float* P = pos + (size_t)cloud * Np * 3;
    float p0x = P[0], p0y = P[1], p0z = P[2];
    for (int j = tid; j < Np; j += 256) {
        float x = P[j * 3 + 0], y = P[j * 3 + 1], z = P[j * 3 + 2];
        px[j] = x; py[j] = y; pz[j] = z;
        float dx = x - p0x, dy = y - p0y, dz = z - p0z;
        dd[j] = dx * dx + dy * dy + dz * dz;
    }
    if (tid == 0) {
        float* c0 = cpos_out + (size_t)cloud * S * 3;
        c0[0] = p0x; c0[1] = p0y; c0[2] = p0z;
    }
    __syncthreads();
    for (int it = 1; it < S; it++) {
        float bv = -1.0f; int bi = 0;
        for (int j = tid; j < Np; j += 256) {
            float v = dd[j];
            if (v > bv) { bv = v; bi = j; }   // ascending j -> first max (matches jnp.argmax)
        }
        rv[tid] = bv; ri[tid] = bi;
        __syncthreads();
        for (int off = 128; off > 0; off >>= 1) {
            if (tid < off) {
                float v2 = rv[tid + off]; int i2 = ri[tid + off];
                if (v2 > rv[tid] || (v2 == rv[tid] && i2 < ri[tid])) { rv[tid] = v2; ri[tid] = i2; }
            }
            __syncthreads();
        }
        int sel = ri[0];
        float sx = px[sel], sy = py[sel], sz = pz[sel];
        if (tid == 0) {
            float* c = cpos_out + ((size_t)cloud * S + it) * 3;
            c[0] = sx; c[1] = sy; c[2] = sz;
        }
        for (int j = tid; j < Np; j += 256) {
            float dx = px[j] - sx, dy = py[j] - sy, dz = pz[j] - sz;
            float nd = dx * dx + dy * dy + dz * dz;
            if (nd < dd[j]) dd[j] = nd;
        }
        __syncthreads();
    }
}

// ---------------- radius top-k: one workgroup per centroid ----------------
__global__ __launch_bounds__(256) void radius_topk_kernel(const float* __restrict__ pos, int Np,
                                                          const float* __restrict__ cpos, int S,
                                                          float r2, int* __restrict__ nbr) {
    extern __shared__ float sm[];
    float* dd = sm;                  // Np
    float* rv = dd + Np;             // 256
    int*   ri = (int*)(rv + 256);    // 256
    int cent = blockIdx.x;
    int cloud = cent / S;
    int tid = threadIdx.x;
    const float* P = pos + (size_t)cloud * Np * 3;
    const float* cp = cpos + (size_t)cent * 3;
    float cx = cp[0], cy = cp[1], cz = cp[2];
    for (int j = tid; j < Np; j += 256) {
        float dx = P[j * 3 + 0] - cx, dy = P[j * 3 + 1] - cy, dz = P[j * 3 + 2] - cz;
        dd[j] = dx * dx + dy * dy + dz * dz;
    }
    __syncthreads();
    int* nb = nbr + (size_t)cent * KNN;
    int firstIdx = 0;
    for (int t = 0; t < KNN; t++) {
        float bv = __builtin_inff(); int bi = Np;
        for (int j = tid; j < Np; j += 256) {
            float v = dd[j];
            if (v < bv) { bv = v; bi = j; }   // ascending j -> smallest index on ties
        }
        rv[tid] = bv; ri[tid] = bi;
        __syncthreads();
        for (int off = 128; off > 0; off >>= 1) {
            if (tid < off) {
                float v2 = rv[tid + off]; int i2 = ri[tid + off];
                if (v2 < rv[tid] || (v2 == rv[tid] && i2 < ri[tid])) { rv[tid] = v2; ri[tid] = i2; }
            }
            __syncthreads();
        }
        float minv = rv[0]; int mini = ri[0];
        if (t == 0) firstIdx = mini;          // always the centroid itself (d=0)
        if (minv <= r2 && mini < Np) {
            if (tid == 0) { nb[t] = mini; dd[mini] = __builtin_inff(); }
        } else {
            if (tid == 0) nb[t] = firstIdx;   // pad with self; duplicate is no-op under max
        }
        __syncthreads();
    }
}

// ---------------- fused WMMA MLP layers ----------------
// Hin: LDS [64][Kp] f16; Wt: global [Cout][Kp] f16 (L2 resident); 256 threads = 8 waves.
template<bool RELU>
__device__ __forceinline__ void mlp_layer(const half_t* Hin, int Kp, const half_t* __restrict__ Wt,
                                          const float* __restrict__ bias, int Cout, half_t* Hout) {
    int wave = threadIdx.x >> 5;
    int lane = threadIdx.x & 31;
    int n = lane & 15, hi = lane >> 4;
    int nt = 4 * (Cout >> 4);
    for (int t = wave; t < nt; t += 8) {
        int m0 = (t & 3) * 16;
        int n0 = (t >> 2) * 16;
        v8f acc = {};
        for (int kb = 0; kb < Kp; kb += 32) {
            v16h a = load_frag_a(Hin, Kp, m0, kb);
            v16h b = load_frag_b(Wt, Kp, n0, kb);
            acc = wmma_f16(a, b, acc);
        }
        float bv = bias[n0 + n];
#pragma unroll
        for (int r = 0; r < 8; r++) {
            float v = acc[r] + bv;
            if (RELU) v = fmaxf(v, 0.0f);
            Hout[(m0 + r + 8 * hi) * Cout + n0 + n] = (half_t)v;
        }
    }
}

// Final layer (no relu) fused with max over the 64 edge rows -> Red[4][Cout]
__device__ __forceinline__ void mlp_layer_max(const half_t* Hin, int Kp, const half_t* __restrict__ Wt,
                                              const float* __restrict__ bias, int Cout, float* Red) {
    int wave = threadIdx.x >> 5;
    int lane = threadIdx.x & 31;
    int n = lane & 15;
    int nt = 4 * (Cout >> 4);
    for (int t = wave; t < nt; t += 8) {
        int m0 = (t & 3) * 16;
        int n0 = (t >> 2) * 16;
        v8f acc = {};
        for (int kb = 0; kb < Kp; kb += 32) {
            v16h a = load_frag_a(Hin, Kp, m0, kb);
            v16h b = load_frag_b(Wt, Kp, n0, kb);
            acc = wmma_f16(a, b, acc);
        }
        float bv = bias[n0 + n];
        float mx = NEG_INF;
#pragma unroll
        for (int r = 0; r < 8; r++) mx = fmaxf(mx, acc[r] + bv);
        mx = fmaxf(mx, __shfl_xor(mx, 16, 32));   // combine M=r and M=r+8 halves
        if (lane < 16) Red[(t & 3) * Cout + n0 + n] = mx;
    }
}

// ---------------- fused SA module: one workgroup per centroid ----------------
template<int CIN, int C1, int C2, int C3, bool HASX>
__global__ __launch_bounds__(256) void sa_kernel(const float* __restrict__ pos, int Np,
                                                 const float* __restrict__ xfeat, int CX,
                                                 const float* __restrict__ cpos,
                                                 const int* __restrict__ nbr,
                                                 const half_t* __restrict__ Wt1, const float* __restrict__ b1,
                                                 const half_t* __restrict__ Wt2, const float* __restrict__ b2,
                                                 const half_t* __restrict__ Wt3, const float* __restrict__ b3,
                                                 float* __restrict__ hout, int S) {
    __shared__ __align__(16) half_t A0[64 * CIN];
    __shared__ __align__(16) half_t H1[64 * C1];
    __shared__ __align__(16) half_t H2[64 * C2];
    __shared__ float Red[4 * C3];
    int cent = blockIdx.x;
    int cloud = cent / S;
    const float* P  = pos + (size_t)cloud * Np * 3;
    const float* X  = HASX ? (xfeat + (size_t)cloud * Np * CX) : nullptr;
    const float* cp = cpos + (size_t)cent * 3;
    const int*   nb = nbr  + (size_t)cent * KNN;
    float cx = cp[0], cy = cp[1], cz = cp[2];
    // warm caches with the weights for this module
    prefetch_buf(Wt1, CIN * C1);
    prefetch_buf(Wt2, C1 * C2);
    prefetch_buf(Wt3, C2 * C3);
    // gather edge features (f32 -> f16, zero-padded to CIN)
    for (int i = threadIdx.x; i < 64 * CIN; i += 256) {
        int e = i / CIN, c = i % CIN;
        int j = nb[e];
        float v = 0.0f;
        if (HASX) {
            if (c < CX)          v = X[(size_t)j * CX + c];
            else if (c < CX + 3) {
                int d = c - CX;
                v = P[j * 3 + d] - (d == 0 ? cx : (d == 1 ? cy : cz));
            }
        } else {
            if (c < 3) v = P[j * 3 + c] - (c == 0 ? cx : (c == 1 ? cy : cz));
        }
        A0[e * CIN + c] = (half_t)v;
    }
    __syncthreads();
    mlp_layer<true>(A0, CIN, Wt1, b1, C1, H1);
    __syncthreads();
    mlp_layer<true>(H1, C1, Wt2, b2, C2, H2);
    __syncthreads();
    mlp_layer_max(H2, C2, Wt3, b3, C3, Red);
    __syncthreads();
    float* ho = hout + (size_t)cent * C3;
    for (int c = threadIdx.x; c < C3; c += 256) {
        float m = fmaxf(fmaxf(Red[c], Red[C3 + c]), fmaxf(Red[2 * C3 + c], Red[3 * C3 + c]));
        ho[c] = m;
    }
}

// ---------------- generic WMMA GEMM for MLP3 (one wave per 16x16 tile) ----------------
template<bool RELU, bool DOMAX>
__global__ __launch_bounds__(128) void gemm_kernel(const half_t* __restrict__ A, int Kp,
                                                   const half_t* __restrict__ Wt,
                                                   const float* __restrict__ bias,
                                                   int Cout, int rowsTotal,
                                                   half_t* __restrict__ Out,
                                                   float* __restrict__ OutMax, int rowsPerCloud) {
    int wave = threadIdx.x >> 5;
    int lane = threadIdx.x & 31;
    int rowTiles = rowsTotal >> 4;
    int tiles_n = Cout >> 4;
    int tile = blockIdx.x * 4 + wave;
    if (tile >= rowTiles * tiles_n) return;
    int rowTile = tile % rowTiles;
    int colTile = tile / rowTiles;
    int m0 = rowTile * 16, n0 = colTile * 16;
    v8f acc = {};
    for (int kb = 0; kb < Kp; kb += 32) {
        v16h a = load_frag_a(A, Kp, m0, kb);
        v16h b = load_frag_b(Wt, Kp, n0, kb);
        acc = wmma_f16(a, b, acc);
    }
    int n = lane & 15, hi = lane >> 4;
    float bv = bias[n0 + n];
    if (DOMAX) {
        float mx = NEG_INF;
#pragma unroll
        for (int r = 0; r < 8; r++) mx = fmaxf(mx, acc[r] + bv);
        mx = fmaxf(mx, __shfl_xor(mx, 16, 32));
        if (lane < 16) {
            int cloud = m0 / rowsPerCloud;
            atomicMaxF(&OutMax[(size_t)cloud * Cout + n0 + n], mx);
        }
    } else {
#pragma unroll
        for (int r = 0; r < 8; r++) {
            float v = acc[r] + bv;
            if (RELU) v = fmaxf(v, 0.0f);
            Out[(size_t)(m0 + r + 8 * hi) * Cout + n0 + n] = (half_t)v;
        }
    }
}

// ---------------- host orchestration ----------------
extern "C" void kernel_launch(void* const* d_in, const int* in_sizes, int n_in,
                              void* d_out, int out_size, void* d_ws, size_t ws_size,
                              hipStream_t stream) {
    (void)in_sizes; (void)n_in; (void)out_size; (void)ws_size;
    const float* pos = (const float*)d_in[0];
    const float* W[9]  = { (const float*)d_in[1],  (const float*)d_in[3],  (const float*)d_in[5],
                           (const float*)d_in[7],  (const float*)d_in[9],  (const float*)d_in[11],
                           (const float*)d_in[13], (const float*)d_in[15], (const float*)d_in[17] };
    const float* bias[9] = { (const float*)d_in[2],  (const float*)d_in[4],  (const float*)d_in[6],
                             (const float*)d_in[8],  (const float*)d_in[10], (const float*)d_in[12],
                             (const float*)d_in[14], (const float*)d_in[16], (const float*)d_in[18] };
    // layer shapes: {Cin, Cout, KpadTo32}
    const int Cin[9]  = { 3, 64, 64, 131, 128, 128, 259, 256, 512 };
    const int CoutL[9]= { 64, 64, 128, 128, 128, 256, 256, 512, 1024 };
    const int Kp[9]   = { 32, 64, 64, 160, 128, 128, 288, 256, 512 };

    char* ws = (char*)d_ws;
    size_t off = 0;
    auto alloc = [&](size_t bytes) { size_t o = off; off = (off + bytes + 255) & ~(size_t)255; return o; };

    half_t* Wt[9];
    for (int i = 0; i < 9; i++) Wt[i] = (half_t*)(ws + alloc((size_t)CoutL[i] * Kp[i] * sizeof(half_t)));
    float* cpos1 = (float*)(ws + alloc((size_t)BATCH * S1 * 3 * sizeof(float)));
    int*   nbr1  = (int*)  (ws + alloc((size_t)BATCH * S1 * KNN * sizeof(int)));
    float* h1    = (float*)(ws + alloc((size_t)BATCH * S1 * 128 * sizeof(float)));
    float* cpos2 = (float*)(ws + alloc((size_t)BATCH * S2 * 3 * sizeof(float)));
    int*   nbr2  = (int*)  (ws + alloc((size_t)BATCH * S2 * KNN * sizeof(int)));
    float* h2    = (float*)(ws + alloc((size_t)BATCH * S2 * 256 * sizeof(float)));
    half_t* A3   = (half_t*)(ws + alloc((size_t)BATCH * S2 * 288 * sizeof(half_t)));
    half_t* G1   = (half_t*)(ws + alloc((size_t)BATCH * S2 * 256 * sizeof(half_t)));
    half_t* G2   = (half_t*)(ws + alloc((size_t)BATCH * S2 * 512 * sizeof(half_t)));
    float* outF  = (float*)d_out;

    // 1) weights -> f16 transposed+padded
    for (int i = 0; i < 9; i++) {
        int total = CoutL[i] * Kp[i];
        wconv_kernel<<<(total + 255) / 256, 256, 0, stream>>>(W[i], Cin[i], CoutL[i], Kp[i], Wt[i]);
    }
    // 2) init output to -inf for atomic max
    init_out_kernel<<<32, 256, 0, stream>>>(outF, BATCH * 1024);

    // 3) SA1: FPS -> radius -> fused WMMA MLP + max-aggregate
    size_t fpsShm1 = (size_t)(4 * NPTS + 512) * sizeof(float);
    fps_kernel<<<BATCH, 256, fpsShm1, stream>>>(pos, NPTS, S1, cpos1);
    size_t radShm1 = (size_t)(NPTS + 512) * sizeof(float);
    radius_topk_kernel<<<BATCH * S1, 256, radShm1, stream>>>(pos, NPTS, cpos1, S1, 0.2f * 0.2f, nbr1);
    sa_kernel<32, 64, 64, 128, false><<<BATCH * S1, 256, 0, stream>>>(
        pos, NPTS, nullptr, 0, cpos1, nbr1,
        Wt[0], bias[0], Wt[1], bias[1], Wt[2], bias[2], h1, S1);

    // 4) SA2 on (h1, cpos1), N'=S1
    size_t fpsShm2 = (size_t)(4 * S1 + 512) * sizeof(float);
    fps_kernel<<<BATCH, 256, fpsShm2, stream>>>(cpos1, S1, S2, cpos2);
    size_t radShm2 = (size_t)(S1 + 512) * sizeof(float);
    radius_topk_kernel<<<BATCH * S2, 256, radShm2, stream>>>(cpos1, S1, cpos2, S2, 0.4f * 0.4f, nbr2);
    sa_kernel<160, 128, 128, 256, true><<<BATCH * S2, 256, 0, stream>>>(
        cpos1, S1, h1, 128, cpos2, nbr2,
        Wt[3], bias[3], Wt[4], bias[4], Wt[5], bias[5], h2, S2);

    // 5) MLP3: concat(h2, cpos2) -> 256 -> 512 -> 1024, fused global max-pool
    {
        int total = BATCH * S2 * 288;
        build_a3_kernel<<<(total + 255) / 256, 256, 0, stream>>>(h2, cpos2, A3);
    }
    int rows = BATCH * S2;                 // 2048
    int rowTiles = rows / 16;              // 128
    {
        int tiles = rowTiles * (256 / 16); // 2048
        gemm_kernel<true, false><<<(tiles + 3) / 4, 128, 0, stream>>>(
            A3, 288, Wt[6], bias[6], 256, rows, G1, nullptr, 0);
    }
    {
        int tiles = rowTiles * (512 / 16); // 4096
        gemm_kernel<true, false><<<(tiles + 3) / 4, 128, 0, stream>>>(
            G1, 256, Wt[7], bias[7], 512, rows, G2, nullptr, 0);
    }
    {
        int tiles = rowTiles * (1024 / 16); // 8192
        gemm_kernel<false, true><<<(tiles + 3) / 4, 128, 0, stream>>>(
            G2, 512, Wt[8], bias[8], 1024, rows, nullptr, outF, S2);
    }
}
